// HierarchicalGRUBaseline_80805514707180
// MI455X (gfx1250) — matrix operand order repaired and checked by reading, not compiled, not run on hardware
//
#include <hip/hip_runtime.h>
#include <math.h>

// ---------------------------------------------------------------------------
// Hierarchical GRU (worker + planner) for MI455X / gfx1250, wave32 WMMA bf16.
// B=256 rows split into 16 independent workgroups (16 rows each) -> no global
// sync; recurrence state lives in LDS, all GEMMs via v_wmma_f32_16x16x32_bf16.
// K-loops are kept rolled (unroll(disable)) so weight tiles stream from the
// shared L2-resident bf16 weight images instead of being hoisted + spilled.
// ---------------------------------------------------------------------------

constexpr int Bsz  = 256;
constexpr int Ssz  = 512;
constexpr int Edim = 256;
constexpr int Hdim = 512;   // HW == HP == F == 512
constexpr int Cn   = 10;
constexpr int KW   = Edim + Hdim;   // 768, worker input dim
constexpr int HSTR = Hdim + 8;      // padded LDS row stride (bf16 elems)
constexpr int XSTR = Edim + 8;
constexpr int SSTR = 2 * Hdim + 8;  // concat(hw,hp) stride in head kernel

typedef __attribute__((ext_vector_type(16))) __bf16 bf16x16;
typedef __attribute__((ext_vector_type(8)))  float  f32x8;

__device__ __forceinline__ unsigned short f2bf(float f) {
  unsigned int u = __float_as_uint(f);
  u += 0x7FFFu + ((u >> 16) & 1u);          // round-to-nearest-even
  return (unsigned short)(u >> 16);
}
__device__ __forceinline__ float bf2f(unsigned short h) {
  return __uint_as_float(((unsigned int)h) << 16);
}
__device__ __forceinline__ float sigmf(float x) { return 1.0f / (1.0f + __expf(-x)); }

__device__ __forceinline__ f32x8 wmma_bf16(bf16x16 a, bf16x16 b, f32x8 c) {
  return __builtin_amdgcn_wmma_f32_16x16x32_bf16(false, a, false, b, (short)0, c,
                                                 false, false);
}

// A operand (16x32, MxK) from row-major [16][ld] bf16 buffer.
// ISA layout: lane m=L&15; lanes 0-15 hold K 0-7 & 16-23, lanes 16-31 hold 8-15 & 24-31.
__device__ __forceinline__ bf16x16 load_a(const unsigned short* base, int ld, int lane) {
  const unsigned short* p = base + (lane & 15) * ld + ((lane >> 4) << 3);
  bf16x16 r;
  __builtin_memcpy(&r, p, 16);
  __builtin_memcpy(reinterpret_cast<char*>(&r) + 16, p + 16, 16);
  return r;
}
// B operand (32x16, KxN) from weights stored row-major W[n][k] with leading dim ldk.
// ISA layout: lane n=L&15; lanes 0-15 hold K 0-7 & 16-23, lanes 16-31 hold 8-15 & 24-31.
__device__ __forceinline__ bf16x16 load_b(const unsigned short* W, int ldk,
                                          int n0, int k0, int lane) {
  const unsigned short* p =
      W + (size_t)(n0 + (lane & 15)) * (size_t)ldk + k0 + ((lane >> 4) << 3);
  bf16x16 r;
  __builtin_memcpy(&r, p, 16);
  __builtin_memcpy(reinterpret_cast<char*>(&r) + 16, p + 16, 16);
  return r;
}

// GRU gate combine for one 16x16 output tile (D-layout: lane n=L&15, m=vgpr+8*(L>>4)).
__device__ __forceinline__ void gru_update(
    const f32x8& ir, const f32x8& iz, const f32x8& inn,
    const f32x8& hr, const f32x8& hz, const f32x8& hn,
    int nb, int lane, int t,
    const float* __restrict__ bi, const float* __restrict__ bh,
    const unsigned short* __restrict__ holdb, unsigned short* __restrict__ hnewb,
    const int* __restrict__ lenb) {
  const int n = nb + (lane & 15);
  const float bir = bi[n],            bhr = bh[n];
  const float biz = bi[Hdim + n],     bhz = bh[Hdim + n];
  const float bin = bi[2 * Hdim + n], bhn = bh[2 * Hdim + n];
  const int mb = (lane >> 4) * 8;
#pragma unroll
  for (int v = 0; v < 8; ++v) {
    const int m = mb + v;
    const float r  = sigmf(ir[v] + bir + hr[v] + bhr);
    const float z  = sigmf(iz[v] + biz + hz[v] + bhz);
    const float nn = tanhf(inn[v] + bin + r * (hn[v] + bhn));
    const float hold = bf2f(holdb[m * HSTR + n]);
    const float hnew = (1.0f - z) * nn + z * hold;
    hnewb[m * HSTR + n] = f2bf((t < lenb[m]) ? hnew : hold);
  }
}

// ---------------------------------------------------------------------------
__global__ void cvt_bf16_kernel(const float* __restrict__ src,
                                unsigned short* __restrict__ dst, int n) {
  int i = blockIdx.x * blockDim.x + threadIdx.x;
  if (i < n) dst[i] = f2bf(src[i]);
}

// ---------------------------------------------------------------------------
// Recurrent kernel: one workgroup per 16 batch rows, 512 threads = 16 waves.
// Each wave owns hidden-column tiles {wave, wave+16} (32 tiles of 16 cols).
// ---------------------------------------------------------------------------
__global__ void __launch_bounds__(512, 1)
hgru_rnn_kernel(const int* __restrict__ ids, const int* __restrict__ lengths,
                const float* __restrict__ emb,
                const unsigned short* __restrict__ WihW,
                const unsigned short* __restrict__ WhhW,
                const float* __restrict__ biW, const float* __restrict__ bhW,
                const unsigned short* __restrict__ Pin,
                const float* __restrict__ pinb,
                const unsigned short* __restrict__ WihP,
                const unsigned short* __restrict__ WhhP,
                const float* __restrict__ biP, const float* __restrict__ bhP,
                float* __restrict__ hw_out, float* __restrict__ hp_out) {
  extern __shared__ unsigned short smem[];
  unsigned short* hwb = smem;                    // [2][16][HSTR] worker state (double buf)
  unsigned short* hpb = hwb + 2 * 16 * HSTR;     // [2][16][HSTR] planner state
  unsigned short* xb  = hpb + 2 * 16 * HSTR;     // [16][XSTR] embedded token tile
  unsigned short* pib = xb + 16 * XSTR;          // [16][HSTR] planner input pi
  int* lenb = (int*)(pib + 16 * HSTR);           // [16]

  const int tid = threadIdx.x, lane = tid & 31, wave = tid >> 5;
  const int rowBase = blockIdx.x * 16;

  for (int i = tid; i < 2 * 16 * HSTR; i += 512) { hwb[i] = 0; hpb[i] = 0; }
  if (tid < 16) lenb[tid] = lengths[rowBase + tid];
  __syncthreads();

  int cw = 0, cp = 0;
  const int n0 = wave * 16, n1 = (wave + 16) * 16;

  for (int t = 0; t < Ssz; ++t) {
    { // embedding gather: row = wave, 8 cols per lane, fp32 -> bf16
      const int id = ids[(rowBase + wave) * Ssz + t];
      unsigned short* dst = xb + wave * XSTR + lane * 8;
      if (id == 0) {
#pragma unroll
        for (int i = 0; i < 8; ++i) dst[i] = 0;   // padding_idx = 0
      } else {
        const float* src = emb + (size_t)id * Edim + lane * 8;
#pragma unroll
        for (int i = 0; i < 8; ++i) dst[i] = f2bf(src[i]);
      }
    }
    __syncthreads();

    const unsigned short* hwcur = hwb + cw * 16 * HSTR;
    const unsigned short* hpcur = hpb + cp * 16 * HSTR;

    // Worker: gi = [xt,hp] @ WihW^T (K=768), gh = hw @ WhhW^T (K=512)
    f32x8 ir0 = {}, iz0 = {}, in0 = {}, ir1 = {}, iz1 = {}, in1 = {};
    f32x8 hr0 = {}, hz0 = {}, hn0 = {}, hr1 = {}, hz1 = {}, hn1 = {};

    // K range [0,256): A from embedded tokens; gh also active.
#pragma clang loop unroll(disable)
    for (int kk = 0; kk < 8; ++kk) {
      const int k0 = kk * 32;
      bf16x16 a = load_a(xb + k0, XSTR, lane);
      ir0 = wmma_bf16(a, load_b(WihW, KW, 0 * Hdim + n0, k0, lane), ir0);
      iz0 = wmma_bf16(a, load_b(WihW, KW, 1 * Hdim + n0, k0, lane), iz0);
      in0 = wmma_bf16(a, load_b(WihW, KW, 2 * Hdim + n0, k0, lane), in0);
      ir1 = wmma_bf16(a, load_b(WihW, KW, 0 * Hdim + n1, k0, lane), ir1);
      iz1 = wmma_bf16(a, load_b(WihW, KW, 1 * Hdim + n1, k0, lane), iz1);
      in1 = wmma_bf16(a, load_b(WihW, KW, 2 * Hdim + n1, k0, lane), in1);
      bf16x16 ah = load_a(hwcur + k0, HSTR, lane);
      hr0 = wmma_bf16(ah, load_b(WhhW, Hdim, 0 * Hdim + n0, k0, lane), hr0);
      hz0 = wmma_bf16(ah, load_b(WhhW, Hdim, 1 * Hdim + n0, k0, lane), hz0);
      hn0 = wmma_bf16(ah, load_b(WhhW, Hdim, 2 * Hdim + n0, k0, lane), hn0);
      hr1 = wmma_bf16(ah, load_b(WhhW, Hdim, 0 * Hdim + n1, k0, lane), hr1);
      hz1 = wmma_bf16(ah, load_b(WhhW, Hdim, 1 * Hdim + n1, k0, lane), hz1);
      hn1 = wmma_bf16(ah, load_b(WhhW, Hdim, 2 * Hdim + n1, k0, lane), hn1);
    }
    // K range [256,512): A from hp; gh still active.
#pragma clang loop unroll(disable)
    for (int kk = 8; kk < 16; ++kk) {
      const int k0 = kk * 32;
      bf16x16 a = load_a(hpcur + (k0 - Edim), HSTR, lane);
      ir0 = wmma_bf16(a, load_b(WihW, KW, 0 * Hdim + n0, k0, lane), ir0);
      iz0 = wmma_bf16(a, load_b(WihW, KW, 1 * Hdim + n0, k0, lane), iz0);
      in0 = wmma_bf16(a, load_b(WihW, KW, 2 * Hdim + n0, k0, lane), in0);
      ir1 = wmma_bf16(a, load_b(WihW, KW, 0 * Hdim + n1, k0, lane), ir1);
      iz1 = wmma_bf16(a, load_b(WihW, KW, 1 * Hdim + n1, k0, lane), iz1);
      in1 = wmma_bf16(a, load_b(WihW, KW, 2 * Hdim + n1, k0, lane), in1);
      bf16x16 ah = load_a(hwcur + k0, HSTR, lane);
      hr0 = wmma_bf16(ah, load_b(WhhW, Hdim, 0 * Hdim + n0, k0, lane), hr0);
      hz0 = wmma_bf16(ah, load_b(WhhW, Hdim, 1 * Hdim + n0, k0, lane), hz0);
      hn0 = wmma_bf16(ah, load_b(WhhW, Hdim, 2 * Hdim + n0, k0, lane), hn0);
      hr1 = wmma_bf16(ah, load_b(WhhW, Hdim, 0 * Hdim + n1, k0, lane), hr1);
      hz1 = wmma_bf16(ah, load_b(WhhW, Hdim, 1 * Hdim + n1, k0, lane), hz1);
      hn1 = wmma_bf16(ah, load_b(WhhW, Hdim, 2 * Hdim + n1, k0, lane), hn1);
    }
    // K range [512,768): A from hp; gi only.
#pragma clang loop unroll(disable)
    for (int kk = 16; kk < 24; ++kk) {
      const int k0 = kk * 32;
      bf16x16 a = load_a(hpcur + (k0 - Edim), HSTR, lane);
      ir0 = wmma_bf16(a, load_b(WihW, KW, 0 * Hdim + n0, k0, lane), ir0);
      iz0 = wmma_bf16(a, load_b(WihW, KW, 1 * Hdim + n0, k0, lane), iz0);
      in0 = wmma_bf16(a, load_b(WihW, KW, 2 * Hdim + n0, k0, lane), in0);
      ir1 = wmma_bf16(a, load_b(WihW, KW, 0 * Hdim + n1, k0, lane), ir1);
      iz1 = wmma_bf16(a, load_b(WihW, KW, 1 * Hdim + n1, k0, lane), iz1);
      in1 = wmma_bf16(a, load_b(WihW, KW, 2 * Hdim + n1, k0, lane), in1);
    }

    unsigned short* hwn = hwb + (cw ^ 1) * 16 * HSTR;
    gru_update(ir0, iz0, in0, hr0, hz0, hn0, n0, lane, t, biW, bhW, hwcur, hwn, lenb);
    gru_update(ir1, iz1, in1, hr1, hz1, hn1, n1, lane, t, biW, bhW, hwcur, hwn, lenb);
    __syncthreads();
    cw ^= 1;

    if (((t + 1) & 3) == 0) {  // planner interval
      const unsigned short* hwc = hwb + cw * 16 * HSTR;
      // pi = hw @ Pin^T + pin_b  (K=512)
      f32x8 p0 = {}, p1 = {};
#pragma clang loop unroll(disable)
      for (int kk = 0; kk < 16; ++kk) {
        const int k0 = kk * 32;
        bf16x16 a = load_a(hwc + k0, HSTR, lane);
        p0 = wmma_bf16(a, load_b(Pin, Hdim, n0, k0, lane), p0);
        p1 = wmma_bf16(a, load_b(Pin, Hdim, n1, k0, lane), p1);
      }
      {
        const int cA = n0 + (lane & 15), cB = n1 + (lane & 15);
        const float bA = pinb[cA], bB = pinb[cB];
        const int mb = (lane >> 4) * 8;
#pragma unroll
        for (int v = 0; v < 8; ++v) {
          pib[(mb + v) * HSTR + cA] = f2bf(p0[v] + bA);
          pib[(mb + v) * HSTR + cB] = f2bf(p1[v] + bB);
        }
      }
      __syncthreads();

      // planner GRU: gi_p = pi @ WihP^T, gh_p = hp @ WhhP^T  (both K=512)
      f32x8 qr0 = {}, qz0 = {}, qn0 = {}, qr1 = {}, qz1 = {}, qn1 = {};
      f32x8 sr0 = {}, sz0 = {}, sn0 = {}, sr1 = {}, sz1 = {}, sn1 = {};
#pragma clang loop unroll(disable)
      for (int kk = 0; kk < 16; ++kk) {
        const int k0 = kk * 32;
        bf16x16 a = load_a(pib + k0, HSTR, lane);
        qr0 = wmma_bf16(a, load_b(WihP, Hdim, 0 * Hdim + n0, k0, lane), qr0);
        qz0 = wmma_bf16(a, load_b(WihP, Hdim, 1 * Hdim + n0, k0, lane), qz0);
        qn0 = wmma_bf16(a, load_b(WihP, Hdim, 2 * Hdim + n0, k0, lane), qn0);
        qr1 = wmma_bf16(a, load_b(WihP, Hdim, 0 * Hdim + n1, k0, lane), qr1);
        qz1 = wmma_bf16(a, load_b(WihP, Hdim, 1 * Hdim + n1, k0, lane), qz1);
        qn1 = wmma_bf16(a, load_b(WihP, Hdim, 2 * Hdim + n1, k0, lane), qn1);
        bf16x16 ah = load_a(hpcur + k0, HSTR, lane);
        sr0 = wmma_bf16(ah, load_b(WhhP, Hdim, 0 * Hdim + n0, k0, lane), sr0);
        sz0 = wmma_bf16(ah, load_b(WhhP, Hdim, 1 * Hdim + n0, k0, lane), sz0);
        sn0 = wmma_bf16(ah, load_b(WhhP, Hdim, 2 * Hdim + n0, k0, lane), sn0);
        sr1 = wmma_bf16(ah, load_b(WhhP, Hdim, 0 * Hdim + n1, k0, lane), sr1);
        sz1 = wmma_bf16(ah, load_b(WhhP, Hdim, 1 * Hdim + n1, k0, lane), sz1);
        sn1 = wmma_bf16(ah, load_b(WhhP, Hdim, 2 * Hdim + n1, k0, lane), sn1);
      }
      unsigned short* hpn = hpb + (cp ^ 1) * 16 * HSTR;
      gru_update(qr0, qz0, qn0, sr0, sz0, sn0, n0, lane, t, biP, bhP, hpcur, hpn, lenb);
      gru_update(qr1, qz1, qn1, sr1, sz1, sn1, n1, lane, t, biP, bhP, hpcur, hpn, lenb);
      __syncthreads();
      cp ^= 1;
    }
  }

  const unsigned short* hwf = hwb + cw * 16 * HSTR;
  const unsigned short* hpf = hpb + cp * 16 * HSTR;
  for (int i = tid; i < 16 * Hdim; i += 512) {
    const int m = i >> 9, c = i & (Hdim - 1);
    hw_out[(size_t)(rowBase + m) * Hdim + c] = bf2f(hwf[m * HSTR + c]);
    hp_out[(size_t)(rowBase + m) * Hdim + c] = bf2f(hpf[m * HSTR + c]);
  }
}

// ---------------------------------------------------------------------------
// Head: fused = tanh([hw|hp]@Fus^T + b); h = relu(fused@H1^T + b); out = h@H2^T + b
// ---------------------------------------------------------------------------
__global__ void __launch_bounds__(512, 1)
hgru_head_kernel(const float* __restrict__ hw_out, const float* __restrict__ hp_out,
                 const unsigned short* __restrict__ Fus, const float* __restrict__ fusb,
                 const unsigned short* __restrict__ H1,  const float* __restrict__ h1b,
                 const float* __restrict__ h2w, const float* __restrict__ h2b,
                 float* __restrict__ out) {
  extern __shared__ unsigned short smem[];
  unsigned short* sb = smem;               // [16][SSTR] concat(hw,hp) bf16
  unsigned short* fb = sb + 16 * SSTR;     // [16][HSTR] fused
  unsigned short* hb = fb + 16 * HSTR;     // [16][HSTR] h1 output

  const int tid = threadIdx.x, lane = tid & 31, wave = tid >> 5;
  const int rowBase = blockIdx.x * 16;

  for (int i = tid; i < 16 * 2 * Hdim; i += 512) {
    const int m = i >> 10, c = i & (2 * Hdim - 1);
    const float v = (c < Hdim) ? hw_out[(size_t)(rowBase + m) * Hdim + c]
                               : hp_out[(size_t)(rowBase + m) * Hdim + (c - Hdim)];
    sb[m * SSTR + c] = f2bf(v);
  }
  __syncthreads();

  const int n0 = wave * 16, n1 = (wave + 16) * 16;
  const int cA = n0 + (lane & 15), cB = n1 + (lane & 15);
  const int mb = (lane >> 4) * 8;

  // fused (K = 1024)
  f32x8 f0 = {}, f1 = {};
#pragma clang loop unroll(disable)
  for (int kk = 0; kk < 32; ++kk) {
    const int k0 = kk * 32;
    bf16x16 a = load_a(sb + k0, SSTR, lane);
    f0 = wmma_bf16(a, load_b(Fus, 2 * Hdim, n0, k0, lane), f0);
    f1 = wmma_bf16(a, load_b(Fus, 2 * Hdim, n1, k0, lane), f1);
  }
  {
    const float bA = fusb[cA], bB = fusb[cB];
#pragma unroll
    for (int v = 0; v < 8; ++v) {
      fb[(mb + v) * HSTR + cA] = f2bf(tanhf(f0[v] + bA));
      fb[(mb + v) * HSTR + cB] = f2bf(tanhf(f1[v] + bB));
    }
  }
  __syncthreads();

  // h1 + relu (K = 512)
  f32x8 g0 = {}, g1 = {};
#pragma clang loop unroll(disable)
  for (int kk = 0; kk < 16; ++kk) {
    const int k0 = kk * 32;
    bf16x16 a = load_a(fb + k0, HSTR, lane);
    g0 = wmma_bf16(a, load_b(H1, Hdim, n0, k0, lane), g0);
    g1 = wmma_bf16(a, load_b(H1, Hdim, n1, k0, lane), g1);
  }
  {
    const float bA = h1b[cA], bB = h1b[cB];
#pragma unroll
    for (int v = 0; v < 8; ++v) {
      float a0 = g0[v] + bA, a1 = g1[v] + bB;
      hb[(mb + v) * HSTR + cA] = f2bf(a0 > 0.0f ? a0 : 0.0f);
      hb[(mb + v) * HSTR + cB] = f2bf(a1 > 0.0f ? a1 : 0.0f);
    }
  }
  __syncthreads();

  // final tiny projection to C=10 classes (scalar fp32)
  if (tid < 16 * Cn) {
    const int m = tid / Cn, c = tid % Cn;
    float acc = h2b[c];
#pragma clang loop unroll(disable)
    for (int k = 0; k < Hdim; ++k) acc += bf2f(hb[m * HSTR + k]) * h2w[c * Hdim + k];
    out[(size_t)(rowBase + m) * Cn + c] = acc;
  }
}

// ---------------------------------------------------------------------------
// Workspace layout (bytes) and launch.
// ---------------------------------------------------------------------------
constexpr size_t WIHW_OFF = 0;                                       // 1536x768 bf16
constexpr size_t WHHW_OFF = WIHW_OFF + (size_t)1536 * 768 * 2;       // 1536x512 bf16
constexpr size_t PIN_OFF  = WHHW_OFF + (size_t)1536 * 512 * 2;       // 512x512 bf16
constexpr size_t WIHP_OFF = PIN_OFF  + (size_t)512 * 512 * 2;        // 1536x512 bf16
constexpr size_t WHHP_OFF = WIHP_OFF + (size_t)1536 * 512 * 2;       // 1536x512 bf16
constexpr size_t FUS_OFF  = WHHP_OFF + (size_t)1536 * 512 * 2;       // 512x1024 bf16
constexpr size_t H1_OFF   = FUS_OFF  + (size_t)512 * 1024 * 2;       // 512x512 bf16
constexpr size_t HWO_OFF  = H1_OFF   + (size_t)512 * 512 * 2;        // 256x512 f32
constexpr size_t HPO_OFF  = HWO_OFF  + (size_t)256 * 512 * 4;        // 256x512 f32

constexpr unsigned RNN_SMEM  = (unsigned)((2*16*HSTR + 2*16*HSTR + 16*XSTR + 16*HSTR) * 2 + 16 * 4);
constexpr unsigned HEAD_SMEM = (unsigned)((16*SSTR + 16*HSTR + 16*HSTR) * 2);

extern "C" void kernel_launch(void* const* d_in, const int* in_sizes, int n_in,
                              void* d_out, int out_size, void* d_ws, size_t ws_size,
                              hipStream_t stream) {
  (void)in_sizes; (void)n_in; (void)out_size; (void)ws_size;
  const int*   ids  = (const int*)  d_in[0];
  const int*   lens = (const int*)  d_in[1];
  const float* emb  = (const float*)d_in[2];
  const float* wihw = (const float*)d_in[3];
  const float* whhw = (const float*)d_in[4];
  const float* bihw = (const float*)d_in[5];
  const float* bhhw = (const float*)d_in[6];
  const float* pinw = (const float*)d_in[7];
  const float* pinb = (const float*)d_in[8];
  const float* wihp = (const float*)d_in[9];
  const float* whhp = (const float*)d_in[10];
  const float* bihp = (const float*)d_in[11];
  const float* bhhp = (const float*)d_in[12];
  const float* fusw = (const float*)d_in[13];
  const float* fusb = (const float*)d_in[14];
  const float* h1w  = (const float*)d_in[15];
  const float* h1b  = (const float*)d_in[16];
  const float* h2w  = (const float*)d_in[17];
  const float* h2b  = (const float*)d_in[18];
  float* out = (float*)d_out;

  char* ws = (char*)d_ws;
  unsigned short* WihW = (unsigned short*)(ws + WIHW_OFF);
  unsigned short* WhhW = (unsigned short*)(ws + WHHW_OFF);
  unsigned short* Pin  = (unsigned short*)(ws + PIN_OFF);
  unsigned short* WihP = (unsigned short*)(ws + WIHP_OFF);
  unsigned short* WhhP = (unsigned short*)(ws + WHHP_OFF);
  unsigned short* Fus  = (unsigned short*)(ws + FUS_OFF);
  unsigned short* H1w  = (unsigned short*)(ws + H1_OFF);
  float* hwo = (float*)(ws + HWO_OFF);
  float* hpo = (float*)(ws + HPO_OFF);

  auto cvt = [&](const float* s, unsigned short* d, int n) {
    cvt_bf16_kernel<<<(n + 255) / 256, 256, 0, stream>>>(s, d, n);
  };
  cvt(wihw, WihW, 1536 * 768);
  cvt(whhw, WhhW, 1536 * 512);
  cvt(pinw, Pin,  512 * 512);
  cvt(wihp, WihP, 1536 * 512);
  cvt(whhp, WhhP, 1536 * 512);
  cvt(fusw, Fus,  512 * 1024);
  cvt(h1w,  H1w,  512 * 512);

  (void)hipFuncSetAttribute((const void*)hgru_rnn_kernel,
                            hipFuncAttributeMaxDynamicSharedMemorySize, RNN_SMEM);
  (void)hipFuncSetAttribute((const void*)hgru_head_kernel,
                            hipFuncAttributeMaxDynamicSharedMemorySize, HEAD_SMEM);

  hgru_rnn_kernel<<<Bsz / 16, 512, RNN_SMEM, stream>>>(
      ids, lens, emb, WihW, WhhW, bihw, bhhw, Pin, pinb, WihP, WhhP, bihp, bhhp,
      hwo, hpo);

  hgru_head_kernel<<<Bsz / 16, 512, HEAD_SMEM, stream>>>(
      hwo, hpo, Fus, fusb, H1w, h1b, h2w, h2b, out);
}
